// MarketStateEstimation_67817533604194
// MI455X (gfx1250) — matrix operand order repaired and checked by reading, not compiled, not run on hardware
//
#include <hip/hip_runtime.h>
#include <hip/hip_bf16.h>
#include <math.h>

typedef __bf16 bf16_t;
typedef __attribute__((ext_vector_type(16))) __bf16 v16bf;
typedef __attribute__((ext_vector_type(8)))  __bf16 v8bf;
typedef __attribute__((ext_vector_type(8)))  float  v8f;

#define B_   8
#define P_   32
#define S_   512
#define D_   64
#define TPE  16
#define XS_STRIDE 40   // bf16 elems per LDS row: 80B, 16B-aligned rows

// ---------------------------------------------------------------------------
// Kernel 1: t-normalization + sinusoidal temporal positional encoding (bf16)
// ---------------------------------------------------------------------------
__global__ __launch_bounds__(512) void tpe_kernel(const float* __restrict__ t,
                                                  const float* __restrict__ alpha_w,
                                                  bf16_t* __restrict__ tpe) {
  __shared__ float smin[512];
  __shared__ float smax[512];
  const int b = blockIdx.x, tid = threadIdx.x;
  float tv = t[b * S_ + tid];
  smin[tid] = tv; smax[tid] = tv;
  __syncthreads();
  for (int off = 256; off > 0; off >>= 1) {
    if (tid < off) {
      smin[tid] = fminf(smin[tid], smin[tid + off]);
      smax[tid] = fmaxf(smax[tid], smax[tid + off]);
    }
    __syncthreads();
  }
  const float mn = smin[0], mx = smax[0];
  float tn = (tv - mn) / (mx - mn);
  const float a = alpha_w[0] * 1000.0f;
  const float scale = (a > 0.0f ? a : (__expf(a) - 1.0f)) + 1.0f;  // elu + 1
  tn *= scale;
  const float c = -2.0f * __logf(10000.0f) / (float)TPE;
  bf16_t* row = tpe + ((size_t)b * S_ + tid) * TPE;
  #pragma unroll
  for (int i = 0; i < TPE / 2; ++i) {
    const float ang = tn * __expf((float)i * c);
    row[2 * i]     = (bf16_t)__sinf(ang);
    row[2 * i + 1] = (bf16_t)__cosf(ang);
  }
}

// ---------------------------------------------------------------------------
// Kernel 2: fused scores + mask + (no-max) softmax + P@X + blend, all WMMA
// One 1024-thread block (32 waves) per (b,p); wave w owns s-rows [16w,16w+16)
// ---------------------------------------------------------------------------
__global__ __launch_bounds__(1024) void attn_kernel(const float* __restrict__ x,
                                                    const float* __restrict__ mask,
                                                    const bf16_t* __restrict__ tpe,
                                                    float* __restrict__ out) {
  __shared__ __align__(16) bf16_t xs[D_ * XS_STRIDE];  // x chunk, [d][t] bf16
  __shared__ float ms[32];                             // mask chunk
  __shared__ float dl[32][16];                         // per-wave denominators

  const int bp   = blockIdx.x;      // b*P + p
  const int b    = bp >> 5;         // P == 32
  const int tid  = threadIdx.x;
  const int wave = tid >> 5;
  const int lane = tid & 31;
  const int hi   = lane >> 4;       // lane-half select
  const int ln   = lane & 15;
  const int s0   = wave * 16;       // this wave's 16 query rows

  const float*  xb = x    + (size_t)bp * S_ * D_;
  const float*  mb = mask + (size_t)bp * S_;
  const bf16_t* tb = tpe  + (size_t)b  * S_ * TPE;
  float*        ob = out  + (size_t)bp * S_ * D_;

  // Loop-invariant B operand for score WMMA: N = s columns, K = tpe dims.
  // Lane half hi holds K=8*hi..8*hi+7 (tpe dims of s0+ln) in elements 0-7;
  // elements 8-15 are the zero padding K=16..31.
  v16bf Bs = {};
  {
    const v8bf bsv = *(const v8bf*)(tb + (size_t)(s0 + ln) * TPE + hi * 8);
    #pragma unroll
    for (int j = 0; j < 8; ++j) Bs[j] = bsv[j];
  }

  v8f acc0 = {}, acc1 = {}, acc2 = {}, acc3 = {};
  const v8f zero8 = {};
  float psum = 0.0f;

  for (int tc = 0; tc < S_; tc += 32) {
    // Stage x[tc:tc+32, :] transposed to [d][t] as bf16 (B-operand ready).
    for (int idx = tid; idx < 32 * D_; idx += 1024) {
      const int tt = idx >> 6, d = idx & 63;
      xs[d * XS_STRIDE + tt] = (bf16_t)xb[(size_t)(tc + tt) * D_ + d];
    }
    if (tid < 32) ms[tid] = mb[tc + tid];
    __syncthreads();

    if (tc + 32 < S_)
      __builtin_prefetch(xb + (size_t)(tc + 32) * D_ + (size_t)lane * 8, 0, 0);

    // A operands: rows are t of this chunk, K = tpe dims (zero-pad 16..31).
    // Lane half hi picks tpe dims [8*hi, 8*hi+8).
    v16bf At0 = {}, At1 = {};
    {
      const v8bf a0 = *(const v8bf*)(tb + (size_t)(tc + ln) * TPE + hi * 8);
      const v8bf a1 = *(const v8bf*)(tb + (size_t)(tc + 16 + ln) * TPE + hi * 8);
      #pragma unroll
      for (int j = 0; j < 8; ++j) { At0[j] = a0[j]; At1[j] = a1[j]; }
    }

    // scores^T tiles: D layout -> lane = s column, VGPR r = t row (r + 8*hi)
    v8f S0t = __builtin_amdgcn_wmma_f32_16x16x32_bf16(false, At0, false, Bs,
                                                      (short)0, zero8, false, false);
    v8f S1t = __builtin_amdgcn_wmma_f32_16x16x32_bf16(false, At1, false, Bs,
                                                      (short)0, zero8, false, false);

    // Masked exp (no max subtraction: |score/4| <= 4) -> P directly in the
    // bf16 A-operand layout, entirely lane-local.
    const int srow = s0 + ln;
    v16bf Pa;
    #pragma unroll
    for (int r = 0; r < 8; ++r) {
      const int t0 = tc + r + hi * 8;
      const int t1 = t0 + 16;
      float e0 = 0.0f, e1 = 0.0f;
      if (ms[t0 - tc] >= 0.5f || t0 == srow) e0 = __expf(S0t[r] * 0.25f);
      if (ms[t1 - tc] >= 0.5f || t1 == srow) e1 = __expf(S1t[r] * 0.25f);
      psum += e0 + e1;
      Pa[r]     = (bf16_t)e0;   // K = r + 8*hi
      Pa[8 + r] = (bf16_t)e1;   // K = 16 + r + 8*hi
    }

    // P @ X over 4 d-tiles: B operand lane = d column, K = t (half-split).
    {
      const bf16_t* xh = xs + hi * 8;
      v16bf Bx0, Bx1, Bx2, Bx3;
      {
        const v8bf l0 = *(const v8bf*)(xh + (0 * 16 + ln) * XS_STRIDE), u0 = *(const v8bf*)(xh + (0 * 16 + ln) * XS_STRIDE + 16);
        const v8bf l1 = *(const v8bf*)(xh + (1 * 16 + ln) * XS_STRIDE), u1 = *(const v8bf*)(xh + (1 * 16 + ln) * XS_STRIDE + 16);
        const v8bf l2 = *(const v8bf*)(xh + (2 * 16 + ln) * XS_STRIDE), u2 = *(const v8bf*)(xh + (2 * 16 + ln) * XS_STRIDE + 16);
        const v8bf l3 = *(const v8bf*)(xh + (3 * 16 + ln) * XS_STRIDE), u3 = *(const v8bf*)(xh + (3 * 16 + ln) * XS_STRIDE + 16);
        #pragma unroll
        for (int j = 0; j < 8; ++j) {
          Bx0[j] = l0[j]; Bx0[8 + j] = u0[j];
          Bx1[j] = l1[j]; Bx1[8 + j] = u1[j];
          Bx2[j] = l2[j]; Bx2[8 + j] = u2[j];
          Bx3[j] = l3[j]; Bx3[8 + j] = u3[j];
        }
      }
      acc0 = __builtin_amdgcn_wmma_f32_16x16x32_bf16(false, Pa, false, Bx0,
                                                     (short)0, acc0, false, false);
      acc1 = __builtin_amdgcn_wmma_f32_16x16x32_bf16(false, Pa, false, Bx1,
                                                     (short)0, acc1, false, false);
      acc2 = __builtin_amdgcn_wmma_f32_16x16x32_bf16(false, Pa, false, Bx2,
                                                     (short)0, acc2, false, false);
      acc3 = __builtin_amdgcn_wmma_f32_16x16x32_bf16(false, Pa, false, Bx3,
                                                     (short)0, acc3, false, false);
    }
    __syncthreads();   // protect xs/ms before next chunk overwrites them
  }

  // Softmax denominator: lane covers half the keys for s = s0+ln; the other
  // half lives in lane^16. Combine, park per-row values in LDS for writeback.
  const float denom = psum + __shfl_xor(psum, 16, 32);
  if (lane < 16) dl[wave][lane] = denom;
  __syncthreads();

  // Writeback: D layout -> row s = s0 + r + 8*hi, col d = 16*dt + ln.
  #pragma unroll
  for (int r = 0; r < 8; ++r) {
    const int sg = s0 + r + hi * 8;
    const float dn = dl[wave][r + hi * 8];
    const float mv = mb[sg];
    const float w1 = 1.0f - mv;
    const size_t base = (size_t)sg * D_;
    ob[base +  0 + ln] = xb[base +  0 + ln] * mv + (acc0[r] / dn) * w1;
    ob[base + 16 + ln] = xb[base + 16 + ln] * mv + (acc1[r] / dn) * w1;
    ob[base + 32 + ln] = xb[base + 32 + ln] * mv + (acc2[r] / dn) * w1;
    ob[base + 48 + ln] = xb[base + 48 + ln] * mv + (acc3[r] / dn) * w1;
  }
}

// ---------------------------------------------------------------------------
extern "C" void kernel_launch(void* const* d_in, const int* in_sizes, int n_in,
                              void* d_out, int out_size, void* d_ws, size_t ws_size,
                              hipStream_t stream) {
  (void)in_sizes; (void)n_in; (void)out_size; (void)ws_size;
  const float* t       = (const float*)d_in[0];
  const float* x       = (const float*)d_in[1];
  const float* mask    = (const float*)d_in[2];
  const float* alpha_w = (const float*)d_in[3];
  float* out   = (float*)d_out;
  bf16_t* tpe  = (bf16_t*)d_ws;   // B*S*TPE bf16 = 128 KB scratch

  tpe_kernel<<<dim3(B_), dim3(S_), 0, stream>>>(t, alpha_w, tpe);
  attn_kernel<<<dim3(B_ * P_), dim3(1024), 0, stream>>>(x, mask, tpe, out);
}
